// FlowAttention_10316511445595
// MI455X (gfx1250) — matrix operand order repaired and checked by reading, not compiled, not run on hardware
//
#include <hip/hip_runtime.h>

#define EPS 1e-6f

typedef __attribute__((ext_vector_type(16))) _Float16 v16h;
typedef __attribute__((ext_vector_type(8)))  _Float16 v8h;
typedef __attribute__((ext_vector_type(8)))  float    v8f;

constexpr int Nn = 4, Ll = 4096, Hh = 16, Dd = 64;
constexpr int NH = Nn * Hh;          // 64
constexpr int ROWS = Hh * Dd;        // 1024 floats between consecutive l for fixed (n,h)

// workspace layout (float offsets)
constexpr size_t WS_QSUM = 0;                 // NH*64
constexpr size_t WS_KSUM = 4096;              // NH*64
constexpr size_t WS_QN   = 8192;              // NH*64 (zeroed, atomically accumulated)
constexpr size_t WS_KN   = 12288;             // NH*64 (zeroed, atomically accumulated)
constexpr size_t WS_NROW = 16384;             // NH*L
constexpr size_t WS_RSC  = 16384 + 262144;    // NH*L   rowscale = nrow*row_refine
constexpr size_t WS_CREF = 278528 + 262144;   // NH*L   col_refine
constexpr size_t WS_KV   = 540672 + 262144;   // NH*64*64 (zeroed, atomically accumulated)

__device__ __forceinline__ float frcp(float x) { return __builtin_amdgcn_rcpf(x); }
__device__ __forceinline__ float sigm(float x) { return frcp(1.0f + __expf(-x)); }

__device__ __forceinline__ size_t nh_base(int nh) {
  int n = nh >> 4, h = nh & 15;
  return ((size_t)n * Ll * Hh + h) * (size_t)Dd;
}

// A-fragment (16x32 f16): K-base of VGPR pair g, for a given lane-half.
__device__ __forceinline__ int apair_k(int g, int lh) {
  int b = (g < 4) ? (g << 1) : (16 + ((g - 4) << 1));
  return b + (lh << 3);
}

union FragA { v16h v; uint32_t u[8]; };
union FragB { v16h v; v8h h[2]; };

// ---------------- K0: zero accumulator regions ----------------
__global__ void k0_init(float* ws) {
  int i = blockIdx.x * blockDim.x + threadIdx.x;
  if (i < 8192)   ws[WS_QN + i] = 0.0f;   // covers qn + kn
  if (i < 262144) ws[WS_KV + i] = 0.0f;
}

// ---------------- K1: k_sum, q_sum over L ----------------
__global__ void k1_sums(const float* __restrict__ Q, const float* __restrict__ K, float* ws) {
  __shared__ float sQ[256], sK[256];
  int nh = blockIdx.x;
  size_t base = nh_base(nh);
  int t = threadIdx.x, d = t & 63, r = t >> 6;
  float aq = 0.f, ak = 0.f;
  for (int l = r; l < Ll; l += 4) {
    size_t off = base + (size_t)l * ROWS + d;
    aq += sigm(Q[off]);
    ak += sigm(K[off]);
  }
  sQ[t] = aq; sK[t] = ak;
  __syncthreads();
  if (t < 64) {
    ws[WS_QSUM + nh * 64 + t] = sQ[t] + sQ[t + 64] + sQ[t + 128] + sQ[t + 192];
    ws[WS_KSUM + nh * 64 + t] = sK[t] + sK[t + 64] + sK[t + 128] + sK[t + 192];
  }
}

// ---------------- K2: normalizers + weighted sums ----------------
__global__ void k2_norms(const float* __restrict__ Q, const float* __restrict__ K, float* ws) {
  __shared__ float ks_eps[64], qs_eps[64], accQn[64], accKn[64];
  int b = blockIdx.x;
  int nh = b >> 3, seg = b & 7;
  size_t base = nh_base(nh);
  int t = threadIdx.x, lane = t & 31, w = t >> 5;
  if (t < 64) {
    ks_eps[t] = ws[WS_KSUM + nh * 64 + t] + EPS;
    qs_eps[t] = ws[WS_QSUM + nh * 64 + t] + EPS;
    accQn[t] = 0.f; accKn[t] = 0.f;
  }
  __syncthreads();
  int lbase = seg * 512;
  float qn0 = 0.f, qn1 = 0.f, kn0 = 0.f, kn1 = 0.f;
  for (int i = 0; i < 64; i++) {
    int l = lbase + w + 8 * i;
    const float* qr = Q + base + (size_t)l * ROWS;
    float q0 = sigm(qr[lane]), q1 = sigm(qr[lane + 32]);
    float p = (q0 + EPS) * ks_eps[lane] + (q1 + EPS) * ks_eps[lane + 32];
    for (int m = 16; m; m >>= 1) p += __shfl_xor(p, m, 32);
    float nr = frcp(p);
    if (lane == 0) ws[WS_NROW + (size_t)nh * Ll + l] = nr;
    qn0 += q0 * nr; qn1 += q1 * nr;

    const float* kr = K + base + (size_t)l * ROWS;
    float k0 = sigm(kr[lane]), k1 = sigm(kr[lane + 32]);
    float pc = (k0 + EPS) * qs_eps[lane] + (k1 + EPS) * qs_eps[lane + 32];
    for (int m = 16; m; m >>= 1) pc += __shfl_xor(pc, m, 32);
    float nc = frcp(pc);
    kn0 += k0 * nc; kn1 += k1 * nc;
  }
  atomicAdd(&accQn[lane], qn0); atomicAdd(&accQn[lane + 32], qn1);
  atomicAdd(&accKn[lane], kn0); atomicAdd(&accKn[lane + 32], kn1);
  __syncthreads();
  if (t < 64) {
    atomicAdd(&ws[WS_QN + nh * 64 + t], accQn[t]);
    atomicAdd(&ws[WS_KN + nh * 64 + t], accKn[t]);
  }
}

// ---------------- K3: rowscale + softmax col_refine ----------------
__global__ void k3_refine(const float* __restrict__ Q, const float* __restrict__ K, float* ws) {
  __shared__ float colraw[Ll];
  __shared__ float kn_eps[64], qn_eps[64], wred[8];
  __shared__ float bmax_s, bsum_s;
  int nh = blockIdx.x;
  size_t base = nh_base(nh);
  int t = threadIdx.x, lane = t & 31, w = t >> 5;
  if (t < 64) {
    kn_eps[t] = ws[WS_KN + nh * 64 + t] + EPS;
    qn_eps[t] = ws[WS_QN + nh * 64 + t] + EPS;
  }
  __syncthreads();
  float wmax = -3.4e38f;
  for (int i = 0; i < 512; i++) {
    int l = w + 8 * i;
    const float* qr = Q + base + (size_t)l * ROWS;
    float q0 = sigm(qr[lane]), q1 = sigm(qr[lane + 32]);
    float rr = (q0 + EPS) * kn_eps[lane] + (q1 + EPS) * kn_eps[lane + 32];
    for (int m = 16; m; m >>= 1) rr += __shfl_xor(rr, m, 32);
    if (lane == 0) {
      // row_refine = sigmoid(rr * L/S), L==S -> ratio 1
      ws[WS_RSC + (size_t)nh * Ll + l] = ws[WS_NROW + (size_t)nh * Ll + l] * sigm(rr);
    }
    const float* kr = K + base + (size_t)l * ROWS;
    float k0 = sigm(kr[lane]), k1 = sigm(kr[lane + 32]);
    float cr = (k0 + EPS) * qn_eps[lane] + (k1 + EPS) * qn_eps[lane + 32];
    for (int m = 16; m; m >>= 1) cr += __shfl_xor(cr, m, 32);
    if (lane == 0) colraw[l] = cr;
    wmax = fmaxf(wmax, cr);
  }
  if (lane == 0) wred[w] = wmax;
  __syncthreads();
  if (t == 0) {
    float m = wred[0];
    for (int i = 1; i < 8; i++) m = fmaxf(m, wred[i]);
    bmax_s = m;
  }
  __syncthreads();
  float bm = bmax_s;
  float ps = 0.f;
  for (int s = t; s < Ll; s += 256) ps += __expf(colraw[s] - bm);
  for (int m = 16; m; m >>= 1) ps += __shfl_xor(ps, m, 32);
  __syncthreads();
  if (lane == 0) wred[w] = ps;
  __syncthreads();
  if (t == 0) {
    float sm = 0.f;
    for (int i = 0; i < 8; i++) sm += wred[i];
    bsum_s = sm;
  }
  __syncthreads();
  float scale = (float)Ll * frcp(bsum_s);
  for (int s = t; s < Ll; s += 256)
    ws[WS_CREF + (size_t)nh * Ll + s] = __expf(colraw[s] - bm) * scale;
}

// ---------------- K4: kv[d,e] = sum_s sig(K)[s,d] * (V[s,e]*cref[s])  (WMMA) ----------------
// LDS tiles TRANSPOSED ([d][s], [e][s]) so fragment reads are contiguous.
// Double-buffered LDS + register preload: one barrier/iter, next chunk's global
// loads overlap with the current chunk's ds_loads + WMMAs.
__global__ void k4_kv(const float* __restrict__ K, const float* __restrict__ V, float* ws) {
  __shared__ alignas(16) _Float16 KhT[2][64][40];   // [buf][d][s], row stride 80B
  __shared__ alignas(16) _Float16 VhT[2][64][40];   // [buf][e][s]
  int blk = blockIdx.x;
  int nh = blk >> 2, kseg = blk & 3;
  size_t base = nh_base(nh);
  int t = threadIdx.x, lane = t & 31, w = t >> 5;
  int L0 = lane & 15, lh = lane >> 4;
  int tm = w & 3, tn0 = (w >> 2) * 2;
  v8f acc0 = {}; v8f acc1 = {};
  const float* colref = ws + WS_CREF + (size_t)nh * Ll;
  int lrow = t >> 3, lcol = (t & 7) * 8;

  // preload chunk 0 into registers
  float4 ka, kb, va, vb;
  float cr;
  {
    int s = kseg * 1024 + lrow;
    const float* kr = K + base + (size_t)s * ROWS + lcol;
    const float* vr = V + base + (size_t)s * ROWS + lcol;
    cr = colref[s];
    ka = *(const float4*)kr; kb = *(const float4*)(kr + 4);
    va = *(const float4*)vr; vb = *(const float4*)(vr + 4);
  }

  for (int c = 0; c < 32; c++) {
    int buf = c & 1;
    {
      float kf[8] = {ka.x, ka.y, ka.z, ka.w, kb.x, kb.y, kb.z, kb.w};
      float vf[8] = {va.x, va.y, va.z, va.w, vb.x, vb.y, vb.z, vb.w};
#pragma unroll
      for (int i = 0; i < 8; i++) {
        KhT[buf][lcol + i][lrow] = (_Float16)sigm(kf[i]);
        VhT[buf][lcol + i][lrow] = (_Float16)(vf[i] * cr);
      }
    }
    __syncthreads();
    if (c < 31) {  // issue next chunk's global loads; overlap with ds_loads/WMMA
      int s = kseg * 1024 + (c + 1) * 32 + lrow;
      const float* kr = K + base + (size_t)s * ROWS + lcol;
      const float* vr = V + base + (size_t)s * ROWS + lcol;
      cr = colref[s];
      ka = *(const float4*)kr; kb = *(const float4*)(kr + 4);
      va = *(const float4*)vr; vb = *(const float4*)(vr + 4);
    }
    FragA a;
    int drow = tm * 16 + L0;
#pragma unroll
    for (int g = 0; g < 8; g++)
      a.u[g] = *(const uint32_t*)&KhT[buf][drow][apair_k(g, lh)];
    FragB b0, b1;
    {
      const v8h* p0 = (const v8h*)&VhT[buf][tn0 * 16 + L0][lh * 8];
      const v8h* p1 = (const v8h*)&VhT[buf][(tn0 + 1) * 16 + L0][lh * 8];
      b0.h[0] = p0[0]; b0.h[1] = p0[2];
      b1.h[0] = p1[0]; b1.h[1] = p1[2];
    }
    acc0 = __builtin_amdgcn_wmma_f32_16x16x32_f16(false, a.v, false, b0.v, (short)0, acc0, false, false);
    acc1 = __builtin_amdgcn_wmma_f32_16x16x32_f16(false, a.v, false, b1.v, (short)0, acc1, false, false);
  }
  float* kv = ws + WS_KV + (size_t)nh * 4096;
#pragma unroll
  for (int r = 0; r < 8; r++) {
    int m = tm * 16 + r + lh * 8;
    atomicAdd(&kv[m * 64 + tn0 * 16 + L0], acc0[r]);
    atomicAdd(&kv[m * 64 + (tn0 + 1) * 16 + L0], acc1[r]);
  }
}

// ---------------- K5: x = sig(Q) @ kv, scaled, to [N,L,H,D]  (WMMA) ----------------
__global__ void k5_out(const float* __restrict__ Q, float* __restrict__ out,
                       const float* __restrict__ ws) {
  __shared__ alignas(16) _Float16 Qh[128][72];    // [l][d] row-major (A: pairs contiguous)
  __shared__ alignas(16) _Float16 KVhT[64][72];   // [e][d] transposed (B: two 8-half runs, 16 apart)
  int blk = blockIdx.x;
  int nh = blk >> 5, lseg = blk & 31;
  int l0 = lseg * 128;
  size_t base = nh_base(nh);
  int t = threadIdx.x, lane = t & 31, w = t >> 5;
  int L0 = lane & 15, lh = lane >> 4;
  const float* kv = ws + WS_KV + (size_t)nh * 4096;
  {
    int row = t >> 2, c0 = (t & 3) * 16;   // row = d, c0 = e base
#pragma unroll
    for (int i = 0; i < 16; i += 4) {
      float4 v = *(const float4*)(kv + row * 64 + c0 + i);
      KVhT[c0 + i + 0][row] = (_Float16)v.x;
      KVhT[c0 + i + 1][row] = (_Float16)v.y;
      KVhT[c0 + i + 2][row] = (_Float16)v.z;
      KVhT[c0 + i + 3][row] = (_Float16)v.w;
    }
  }
  {
    int row = t >> 1, c0 = (t & 1) * 32;
    const float* qr = Q + base + (size_t)(l0 + row) * ROWS + c0;
#pragma unroll
    for (int i = 0; i < 32; i += 4) {
      float4 v = *(const float4*)(qr + i);
      Qh[row][c0 + i + 0] = (_Float16)sigm(v.x);
      Qh[row][c0 + i + 1] = (_Float16)sigm(v.y);
      Qh[row][c0 + i + 2] = (_Float16)sigm(v.z);
      Qh[row][c0 + i + 3] = (_Float16)sigm(v.w);
    }
  }
  __syncthreads();
  v8f acc[4] = {{}, {}, {}, {}};
  int mrow = w * 16 + L0;
#pragma unroll
  for (int kc = 0; kc < 64; kc += 32) {
    FragA a;
#pragma unroll
    for (int g = 0; g < 8; g++)
      a.u[g] = *(const uint32_t*)&Qh[mrow][kc + apair_k(g, lh)];
#pragma unroll
    for (int tn = 0; tn < 4; tn++) {
      FragB bf;
      const v8h* p = (const v8h*)&KVhT[tn * 16 + L0][kc + lh * 8];
      bf.h[0] = p[0]; bf.h[1] = p[2];
      acc[tn] = __builtin_amdgcn_wmma_f32_16x16x32_f16(false, a.v, false, bf.v, (short)0, acc[tn], false, false);
    }
  }
  const float* rowscale = ws + WS_RSC + (size_t)nh * Ll;
#pragma unroll
  for (int r = 0; r < 8; r++) {
    int l = l0 + w * 16 + r + lh * 8;
    float rs = rowscale[l];
    float* orow = out + base + (size_t)l * ROWS;
#pragma unroll
    for (int tn = 0; tn < 4; tn++)
      orow[tn * 16 + L0] = acc[tn][r] * rs;
  }
}

extern "C" void kernel_launch(void* const* d_in, const int* in_sizes, int n_in,
                              void* d_out, int out_size, void* d_ws, size_t ws_size,
                              hipStream_t stream) {
  const float* Q = (const float*)d_in[0];
  const float* K = (const float*)d_in[1];
  const float* V = (const float*)d_in[2];
  float* out = (float*)d_out;
  float* ws = (float*)d_ws;

  k0_init<<<1024, 256, 0, stream>>>(ws);
  k1_sums<<<NH, 256, 0, stream>>>(Q, K, ws);
  k2_norms<<<NH * 8, 256, 0, stream>>>(Q, K, ws);
  k3_refine<<<NH, 256, 0, stream>>>(Q, K, ws);
  k4_kv<<<NH * 4, 256, 0, stream>>>(K, V, ws);
  k5_out<<<NH * 32, 256, 0, stream>>>(Q, out, ws);
}
